// GraphLearning_64518998721046
// MI455X (gfx1250) — compile-verified, matrix operand order checked by reading, not run on hardware
//
#include <hip/hip_runtime.h>
#include <hip/hip_bf16.h>

typedef __attribute__((ext_vector_type(16))) _Float16 v16h;
typedef __attribute__((ext_vector_type(8)))  float    v8f;

#define D_CH 128
#define K1   3072
#define BS_  100
#define TOPK_ 50

// ---------------------------------------------------------------------------
// Per-lane async copy of 16 bytes global -> LDS (ASYNCcnt-tracked).
// LDS operand = wave-relative LDS byte address = low 32 bits of the flat
// address of a __shared__ object.
// ---------------------------------------------------------------------------
__device__ __forceinline__ void async_g2l_b128(const void* gptr, void* lptr) {
    unsigned           lds = (unsigned)(size_t)lptr;
    unsigned long long ga  = (unsigned long long)(size_t)gptr;
    asm volatile("global_load_async_to_lds_b128 %0, %1, off"
                 :: "v"(lds), "v"(ga) : "memory");
}
__device__ __forceinline__ void async_wait0() {
    asm volatile("s_wait_asynccnt 0x0" ::: "memory");
}

// ---------------------------------------------------------------------------
// Load a 16x32 f16 fragment (A-layout; B uses same layout on Bt[n][k] tiles)
// per CDNA5 ISA 7.12.2: lanes 0-15 -> M=lane, kBase=0; lanes 16-31 -> kBase=8;
// VGPR v<4: K = kBase+2v ; v>=4: K = kBase+16+2(v-4). Packed f16 pairs -> u32.
// ---------------------------------------------------------------------------
__device__ __forceinline__ v16h load_frag16(const _Float16* base, int stride, int lane) {
    int m  = lane & 15;
    int kb = (lane & 16) ? 8 : 0;
    union { v16h h; unsigned u[8]; } f;
#pragma unroll
    for (int v = 0; v < 8; ++v) {
        int k = ((v < 4) ? (2 * v) : (16 + 2 * (v - 4))) + kb;
        f.u[v] = *(const unsigned*)(base + m * stride + k);
    }
    return f.h;
}

// ---------------------------------------------------------------------------
// Kernel 1: x[N,128] = in[N,3072] @ W1^T + b1   (f16 WMMA, f32 accumulate)
// block = 128 threads (4 waves); tile 64 rows x 128 cols; 96 K-steps of 32.
// (staging must convert f32->f16, so it stays on the load/convert/ds path)
// ---------------------------------------------------------------------------
__global__ void gl_gemm1(const float* __restrict__ in, const float* __restrict__ W1,
                         const float* __restrict__ b1, float* __restrict__ x, int n) {
    __shared__ __align__(16) _Float16 As[64][40];    // 64 x 32 (+8 pad)
    __shared__ __align__(16) _Float16 Bs[128][40];   // Bt[d][k] = W1[d][k0+k]

    const int tid  = threadIdx.x;
    const int wave = tid >> 5;
    const int lane = tid & 31;
    const int row0 = blockIdx.x * 64;

    const v8f vzero = {0.f,0.f,0.f,0.f,0.f,0.f,0.f,0.f};
    v8f acc[8];
#pragma unroll
    for (int i = 0; i < 8; ++i) acc[i] = vzero;

    for (int k0 = 0; k0 < K1; k0 += 32) {
        {   // stage A: 2048 elems, 16 per thread
            int r  = tid >> 1;
            int kk = (tid & 1) * 16;
            int gr = row0 + r;
            if (gr < n) {
                const float* p = in + (size_t)gr * K1 + k0 + kk;
#pragma unroll
                for (int q = 0; q < 16; ++q) As[r][kk + q] = (_Float16)p[q];
            } else {
#pragma unroll
                for (int q = 0; q < 16; ++q) As[r][kk + q] = (_Float16)0.f;
            }
        }
        {   // stage B: thread d loads 32 contiguous f32 of W1 row d
            const float* p = W1 + (size_t)tid * K1 + k0;
#pragma unroll
            for (int q = 0; q < 32; ++q) Bs[tid][q] = (_Float16)p[q];
        }
        __syncthreads();

        v16h afrag = load_frag16(&As[wave * 16][0], 40, lane);
#pragma unroll
        for (int nt = 0; nt < 8; ++nt) {
            v16h bfrag = load_frag16(&Bs[nt * 16][0], 40, lane);
            acc[nt] = __builtin_amdgcn_wmma_f32_16x16x32_f16(
                false, afrag, false, bfrag, (short)0, acc[nt], false, false);
        }
        __syncthreads();
    }

    const int nlo  = lane & 15;
    const int rofs = (lane & 16) ? 8 : 0;
#pragma unroll
    for (int nt = 0; nt < 8; ++nt) {
        int   col  = nt * 16 + nlo;
        float bias = b1[col];
#pragma unroll
        for (int r = 0; r < 8; ++r) {
            int grow = row0 + wave * 16 + r + rofs;
            if (grow < n) x[(size_t)grow * D_CH + col] = acc[nt][r] + bias;
        }
    }
}

// ---------------------------------------------------------------------------
// Kernel 2: per-micro-batch BatchNorm (biased var) + emit f16 GEMM operands
// ---------------------------------------------------------------------------
__global__ void gl_bn(const float* __restrict__ x, const float* __restrict__ gamma,
                      const float* __restrict__ beta, const float* __restrict__ sw,
                      float* __restrict__ outv, _Float16* __restrict__ Uh,
                      _Float16* __restrict__ Vh) {
    const int b = blockIdx.x;       // micro-batch
    const int d = threadIdx.x;      // channel
    const float* xb = x + (size_t)b * BS_ * D_CH + d;

    float s = 0.f, s2 = 0.f;
    for (int r = 0; r < BS_; ++r) {
        float v = xb[(size_t)r * D_CH];
        s += v; s2 += v * v;
    }
    float mean = s * (1.0f / BS_);
    float var  = s2 * (1.0f / BS_) - mean * mean;
    float inv  = rsqrtf(var + 1e-5f);
    float g = gamma[d], be = beta[d], w = sw[d];

    for (int r = 0; r < BS_; ++r) {
        size_t i = (size_t)(b * BS_ + r);
        float  o = (xb[(size_t)r * D_CH] - mean) * inv * g + be;
        outv[i * D_CH + d] = o;
        Uh[i * D_CH + d] = (_Float16)(o * w);
        Vh[i * D_CH + d] = (_Float16)o;
    }
}

// ---------------------------------------------------------------------------
// Kernel 3: a[i] = sum_d sw[d]*out[i,d]^2
// ---------------------------------------------------------------------------
__global__ void gl_arow(const float* __restrict__ outv, const float* __restrict__ sw,
                        float* __restrict__ a) {
    __shared__ float red[D_CH];
    const int i = blockIdx.x, d = threadIdx.x;
    float o = outv[(size_t)i * D_CH + d];
    red[d] = sw[d] * o * o;
    __syncthreads();
    for (int s = D_CH / 2; s > 0; s >>= 1) {
        if (d < s) red[d] += red[d + s];
        __syncthreads();
    }
    if (d == 0) a[i] = red[0];
}

// ---------------------------------------------------------------------------
// Kernel 4: scores = relu(a_i + a_j - 2*(U @ V^T) + sb) into S (scratch)
// block = 128 threads (4 waves); 64x64 tile; K=128 fully staged in LDS via
// GLOBAL_LOAD_ASYNC_TO_LDS_B128 (operands are already f16 in memory).
// ---------------------------------------------------------------------------
__global__ void gl_cross(const _Float16* __restrict__ Uh, const _Float16* __restrict__ Vh,
                         const float* __restrict__ a, const float* __restrict__ sb,
                         float* __restrict__ S, int n) {
    __shared__ __align__(16) _Float16 Us[64][136];   // 64 x 128 (+8 pad)
    __shared__ __align__(16) _Float16 Vs[64][136];

    const int tid  = threadIdx.x;
    const int wave = tid >> 5;
    const int lane = tid & 31;
    const int i0 = blockIdx.y * 64;
    const int j0 = blockIdx.x * 64;

    // stage both tiles: thread -> row r = tid>>1, half c0 = (tid&1)*64
    // 64 halfs = 128 B = 8 async b128 chunks per matrix per thread
    {
        int r  = tid >> 1;
        int c0 = (tid & 1) * 64;
        int gi = i0 + r, gj = j0 + r;
        const float4 z4 = make_float4(0.f, 0.f, 0.f, 0.f);
        if (gi < n) {
            const _Float16* gp = Uh + (size_t)gi * D_CH + c0;
#pragma unroll
            for (int q = 0; q < 8; ++q) async_g2l_b128(gp + 8 * q, &Us[r][c0 + 8 * q]);
        } else {
#pragma unroll
            for (int q = 0; q < 8; ++q) *(float4*)&Us[r][c0 + 8 * q] = z4;
        }
        if (gj < n) {
            const _Float16* gp = Vh + (size_t)gj * D_CH + c0;
#pragma unroll
            for (int q = 0; q < 8; ++q) async_g2l_b128(gp + 8 * q, &Vs[r][c0 + 8 * q]);
        } else {
#pragma unroll
            for (int q = 0; q < 8; ++q) *(float4*)&Vs[r][c0 + 8 * q] = z4;
        }
    }
    async_wait0();
    __syncthreads();

    const v8f vzero = {0.f,0.f,0.f,0.f,0.f,0.f,0.f,0.f};
    v8f acc[4] = {vzero, vzero, vzero, vzero};
#pragma unroll
    for (int k0 = 0; k0 < D_CH; k0 += 32) {
        v16h afrag = load_frag16(&Us[wave * 16][k0], 136, lane);
#pragma unroll
        for (int nt = 0; nt < 4; ++nt) {
            v16h bfrag = load_frag16(&Vs[nt * 16][k0], 136, lane);
            acc[nt] = __builtin_amdgcn_wmma_f32_16x16x32_f16(
                false, afrag, false, bfrag, (short)0, acc[nt], false, false);
        }
    }

    // fused epilogue; scores stream once through HBM -> non-temporal stores
    const float sbv  = *sb;
    const int   nlo  = lane & 15;
    const int   rofs = (lane & 16) ? 8 : 0;
#pragma unroll
    for (int nt = 0; nt < 4; ++nt) {
        int gj = j0 + nt * 16 + nlo;
        if (gj >= n) continue;
        float aj = a[gj];
#pragma unroll
        for (int r = 0; r < 8; ++r) {
            int gi = i0 + wave * 16 + r + rofs;
            if (gi < n) {
                float sc = a[gi] + aj - 2.0f * acc[nt][r] + sbv;
                __builtin_nontemporal_store(sc > 0.f ? sc : 0.f,
                                            &S[(size_t)gi * n + gj]);
            }
        }
    }
}

// ---------------------------------------------------------------------------
// Kernel 5: per-row top-50 -> softmax -> zero row + scatter (in place on S)
// one 256-thread block per row; 40KB row staged in LDS via async b128 DMA.
// ---------------------------------------------------------------------------
__global__ void gl_topk(float* __restrict__ S, int n) {
    __shared__ __align__(16) float vals[10016];
    __shared__ float redV[256];
    __shared__ int   redI[256];
    __shared__ float topV[TOPK_];
    __shared__ int   topI[TOPK_];
    __shared__ float ssum;

    const int row = blockIdx.x;
    const int tid = threadIdx.x;
    float* rowp = S + (size_t)row * n;

    const int chunks = n >> 2;                    // 16B chunks (n % 4 == 0)
    for (int c = tid; c < chunks; c += 256)
        async_g2l_b128(rowp + 4 * c, &vals[4 * c]);
    async_wait0();
    __syncthreads();

    for (int t = 0; t < TOPK_; ++t) {
        float bv = -1e30f; int bi = 0x7fffffff;
        for (int j = tid; j < n; j += 256) {
            float v = vals[j];
            if (v > bv) { bv = v; bi = j; }
        }
        redV[tid] = bv; redI[tid] = bi;
        __syncthreads();
        for (int s = 128; s > 0; s >>= 1) {
            if (tid < s) {
                float ov = redV[tid + s]; int oi = redI[tid + s];
                if (ov > redV[tid] || (ov == redV[tid] && oi < redI[tid])) {
                    redV[tid] = ov; redI[tid] = oi;
                }
            }
            __syncthreads();
        }
        if (tid == 0) {
            topV[t] = redV[0]; topI[t] = redI[0];
            vals[redI[0]] = -1e30f;   // remove winner
        }
        __syncthreads();
    }

    if (tid == 0) {
        float mx = topV[0];          // selection order is descending
        float s = 0.f;
        for (int t = 0; t < TOPK_; ++t) { float e = expf(topV[t] - mx); topV[t] = e; s += e; }
        ssum = s;
    }
    __syncthreads();

    for (int j = tid; j < n; j += 256)
        __builtin_nontemporal_store(0.0f, &rowp[j]);
    __syncthreads();
    if (tid < TOPK_)
        __builtin_nontemporal_store(topV[tid] / ssum, &rowp[topI[tid]]);
}

// ---------------------------------------------------------------------------
extern "C" void kernel_launch(void* const* d_in, const int* in_sizes, int n_in,
                              void* d_out, int out_size, void* d_ws, size_t ws_size,
                              hipStream_t stream) {
    const float* inputs = (const float*)d_in[0];
    const float* W1     = (const float*)d_in[1];
    const float* b1     = (const float*)d_in[2];
    const float* gamma  = (const float*)d_in[3];
    const float* beta   = (const float*)d_in[4];
    const float* sw     = (const float*)d_in[5];
    const float* sb     = (const float*)d_in[6];

    const int N = in_sizes[0] / K1;           // 10000

    float* outv = (float*)d_out;              // [N,128]
    float* S    = outv + (size_t)N * D_CH;    // [N,N] (scratch scores, then final)

    char* ws = (char*)d_ws;
    float*    x  = (float*)ws;     ws += (size_t)N * D_CH * sizeof(float);
    float*    a  = (float*)ws;     ws += (size_t)N * sizeof(float);
    _Float16* Uh = (_Float16*)ws;  ws += (size_t)N * D_CH * sizeof(_Float16);
    _Float16* Vh = (_Float16*)ws;

    const int rb = (N + 63) / 64;

    gl_gemm1<<<rb, 128, 0, stream>>>(inputs, W1, b1, x, N);
    gl_bn<<<N / BS_, D_CH, 0, stream>>>(x, gamma, beta, sw, outv, Uh, Vh);
    gl_arow<<<N, D_CH, 0, stream>>>(outv, sw, a);
    dim3 cg(rb, rb);
    gl_cross<<<cg, 128, 0, stream>>>(Uh, Vh, a, sb, S, N);
    gl_topk<<<N, 256, 0, stream>>>(S, N);
}
